// MIEsitmator_39943195853016
// MI455X (gfx1250) — matrix operand order repaired and checked by reading, not compiled, not run on hardware
//
#include <hip/hip_runtime.h>
#include <hip/hip_bf16.h>
#include <math.h>
#include <stdint.h>

typedef __attribute__((ext_vector_type(16))) _Float16 v16h;
typedef __attribute__((ext_vector_type(8)))  _Float16 v8h;
typedef __attribute__((ext_vector_type(8)))  float    v8f;

#define BATCH 32
#define DMODEL 1024
#define TLEN 1000
#define LMAX 200
#define VOCAB 80
#define BLANK 79
#define NEGF (-1e30f)
#define MROWS (BATCH * TLEN)   // 32000, = 2000 tiles of 16

// ---------------------------------------------------------------- utilities
__global__ void zero_out_kernel(float* out) {
    if (threadIdx.x == 0 && blockIdx.x == 0) out[0] = 0.0f;
}

// decoder_outputs [B][D][T] (f32) -> Xh [B*T][D] (f16).
// Load phase uses GLOBAL_LOAD_ASYNC_TO_LDS_B128 (ASYNCcnt-tracked direct
// global->LDS copy, 16B/lane), drained with s_wait_asynccnt before the
// barrier; write phase is the transposed f16 store.
__global__ __launch_bounds__(256) void transpose_convert_x(
        const float* __restrict__ dec, _Float16* __restrict__ Xh) {
    __shared__ float tile[32][33];
    const int b = blockIdx.z;
    const int tBase = blockIdx.x * 32;
    const int dBase = blockIdx.y * 32;
    const int tid = threadIdx.x;

    {   // 256 threads = 32 d-rows x 8 quads of 4 floats along t
        const int dl = tid >> 3;          // 0..31
        const int q  = tid & 7;           // 0..7
        const int t0 = tBase + q * 4;
        if (t0 + 3 < TLEN) {
            const unsigned lds_off =
                (unsigned)(uintptr_t)&tile[dl][q * 4];   // low 32 bits = LDS byte offset
            const unsigned long long gaddr = (unsigned long long)(uintptr_t)
                (dec + ((size_t)b * DMODEL + dBase + dl) * TLEN + t0);
            asm volatile("global_load_async_to_lds_b128 %0, %1, off"
                         :: "v"(lds_off), "v"(gaddr) : "memory");
        } else {
            tile[dl][q * 4 + 0] = 0.0f;
            tile[dl][q * 4 + 1] = 0.0f;
            tile[dl][q * 4 + 2] = 0.0f;
            tile[dl][q * 4 + 3] = 0.0f;
        }
    }
    asm volatile("s_wait_asynccnt 0x0" ::: "memory");
    __syncthreads();

    const int tx = tid & 31;              // d local on write side
    const int ty = tid >> 5;              // 0..7
#pragma unroll
    for (int j = 0; j < 4; ++j) {
        const int tl = ty * 4 + j;
        const int t = tBase + tl;
        if (t < TLEN)
            Xh[((size_t)b * TLEN + t) * DMODEL + (dBase + tx)] =
                (_Float16)tile[tx][tl];
    }
}

// W1 [K=1024][N=1024] f32 -> W1t [N][K] f16 (column-major B operand)
__global__ __launch_bounds__(256) void convert_w1(
        const float* __restrict__ W1, _Float16* __restrict__ W1t) {
    const int idx = blockIdx.x * 256 + threadIdx.x;   // n*1024 + k
    const int n = idx >> 10;
    const int k = idx & 1023;
    W1t[idx] = (_Float16)W1[(size_t)k * DMODEL + n];
}

// W2 [K=1024][V=80] f32 -> W2t [V][K] f16
__global__ __launch_bounds__(256) void convert_w2(
        const float* __restrict__ W2, _Float16* __restrict__ W2t) {
    const int idx = blockIdx.x * 256 + threadIdx.x;   // v*1024 + k
    if (idx >= VOCAB * DMODEL) return;
    const int v = idx >> 10;
    const int k = idx & 1023;
    W2t[idx] = (_Float16)W2[(size_t)k * VOCAB + v];
}

// ---------------------------------------------------------------- WMMA GEMMs
// Fragment layouts used here (16-bit A 16x32 / B 32x16; contiguous k order, not the ISA table order):
//   A: lane L holds row m=L%16; halves 0..15 =
//      K: (L/16)*16 + 0..15 (one contiguous 32B run)
//   B (stored column-major): lane L holds col n=L%16; halves 0..15 =
//      K: (L/16)*16 + 0..15 (one contiguous 32B run)
__device__ inline v16h load_a_frag(const _Float16* __restrict__ p) {
    const v8h a0 = *(const v8h*)p;
    const v8h a1 = *(const v8h*)(p + 8);
    v16h a;
#pragma unroll
    for (int i = 0; i < 8; ++i) { a[i] = a0[i]; a[8 + i] = a1[i]; }
    return a;
}
__device__ inline v16h load_b_frag(const _Float16* __restrict__ p) {
    const v8h b0 = *(const v8h*)p;
    const v8h b1 = *(const v8h*)(p + 8);
    v16h b;
#pragma unroll
    for (int i = 0; i < 8; ++i) { b[i] = b0[i]; b[8 + i] = b1[i]; }
    return b;
}

// H = relu(Xh @ W1 + b1), f16 out.  Each wave: one M tile x FOUR N tiles
// (A fragment reused by 4 back-to-back WMMAs).  Ping-pong K loop (step 64):
// fragment sets are overwritten directly by the next loads, so no register
// copy chains; ~10 loads stay in flight behind every WMMA group.
// grid(2000, 2), 8 waves/block.
__global__ __launch_bounds__(256) void gemm1_wmma(
        const _Float16* __restrict__ Xh, const _Float16* __restrict__ W1t,
        const float* __restrict__ b1, _Float16* __restrict__ Hh) {
    const int lane = threadIdx.x & 31;
    const int wave = threadIdx.x >> 5;                    // 0..7
    const int tileM = blockIdx.x * 16;
    const int nBase = (blockIdx.y * 8 + wave) * 64;       // 4 N-tiles of 16
    const int nlow = lane & 15;
    const int mrow = tileM + nlow;
    const int ka0 = (lane >> 4) << 4;                     // 0 or 16
    const int kb0 = (lane >> 4) << 4;                     // 0 or 16
    const _Float16* aptr = Xh + (size_t)mrow * DMODEL + ka0;
    const _Float16* bptr = W1t + (size_t)(nBase + nlow) * DMODEL + kb0;

    v8f acc[4] = {{}, {}, {}, {}};
    v16h aA = load_a_frag(aptr);
    v16h bA[4];
#pragma unroll
    for (int j = 0; j < 4; ++j)
        bA[j] = load_b_frag(bptr + (size_t)j * 16 * DMODEL);

    for (int k = 0; k < DMODEL; k += 64) {
        // ---- half 1: load k+32 into set B, compute on set A (data @ k)
        v16h aB = load_a_frag(aptr + k + 32);
        v16h bB[4];
#pragma unroll
        for (int j = 0; j < 4; ++j)
            bB[j] = load_b_frag(bptr + (size_t)j * 16 * DMODEL + k + 32);
        __builtin_prefetch(aptr + k + 256, 0, 3);   // speculative, drop-safe
        __builtin_prefetch(bptr + k + 256, 0, 3);
#pragma unroll
        for (int j = 0; j < 4; ++j)
            acc[j] = __builtin_amdgcn_wmma_f32_16x16x32_f16(
                         false, aA, false, bA[j], (short)0, acc[j],
                         false, false);
        // ---- half 2: reload set A with k+64 (wraps to 0 on last iter),
        //      compute on set B (data @ k+32)
        const int kn = (k + 64) & (DMODEL - 1);
        aA = load_a_frag(aptr + kn);
#pragma unroll
        for (int j = 0; j < 4; ++j)
            bA[j] = load_b_frag(bptr + (size_t)j * 16 * DMODEL + kn);
#pragma unroll
        for (int j = 0; j < 4; ++j)
            acc[j] = __builtin_amdgcn_wmma_f32_16x16x32_f16(
                         false, aB, false, bB[j], (short)0, acc[j],
                         false, false);
    }
    // C layout: VGPR r -> m = (lane/16)*8 + r, n = lane%16
    const int mbase = (lane >> 4) << 3;
#pragma unroll
    for (int j = 0; j < 4; ++j) {
        const int ncol = nBase + j * 16 + nlow;
        const float bias = b1[ncol];
#pragma unroll
        for (int r = 0; r < 8; ++r) {
            float v = acc[j][r] + bias;
            v = v > 0.0f ? v : 0.0f;
            Hh[(size_t)(tileM + mbase + r) * DMODEL + ncol] = (_Float16)v;
        }
    }
}

// logits = Hh @ W2 + b2 fused with row-wise log_softmax over V=80.
// grid(2000), 5 waves/block (V = 5*16); logits staged in LDS, log-probs out.
__global__ __launch_bounds__(160) void gemm2_softmax_wmma(
        const _Float16* __restrict__ Hh, const _Float16* __restrict__ W2t,
        const float* __restrict__ b2, float* __restrict__ LP) {
    __shared__ float lbuf[16][VOCAB + 4];
    const int lane = threadIdx.x & 31;
    const int wave = threadIdx.x >> 5;                    // 0..4
    const int tileM = blockIdx.x * 16;
    const int tileN = wave * 16;
    const int nlow = lane & 15;
    const int mrow = tileM + nlow;
    const int ka0 = (lane >> 4) << 4;
    const int kb0 = (lane >> 4) << 4;
    const _Float16* aptr = Hh + (size_t)mrow * DMODEL + ka0;
    const _Float16* bptr = W2t + (size_t)(tileN + nlow) * DMODEL + kb0;

    v8f acc = {};
    v16h aA = load_a_frag(aptr);
    v16h bA = load_b_frag(bptr);
    for (int k = 0; k < DMODEL; k += 64) {
        v16h aB = load_a_frag(aptr + k + 32);
        v16h bB = load_b_frag(bptr + k + 32);
        __builtin_prefetch(aptr + k + 256, 0, 3);
        __builtin_prefetch(bptr + k + 256, 0, 3);
        acc = __builtin_amdgcn_wmma_f32_16x16x32_f16(
                  false, aA, false, bA, (short)0, acc, false, false);
        const int kn = (k + 64) & (DMODEL - 1);
        aA = load_a_frag(aptr + kn);
        bA = load_b_frag(bptr + kn);
        acc = __builtin_amdgcn_wmma_f32_16x16x32_f16(
                  false, aB, false, bB, (short)0, acc, false, false);
    }
    const float bias = b2[tileN + nlow];
    const int mbase = (lane >> 4) << 3;
#pragma unroll
    for (int r = 0; r < 8; ++r)
        lbuf[mbase + r][tileN + nlow] = acc[r] + bias;
    __syncthreads();

    // log-softmax: each wave handles rows wave, wave+5, wave+10, (wave+15)
    for (int row = wave; row < 16; row += 5) {
        const float x0 = lbuf[row][lane];
        const float x1 = lbuf[row][lane + 32];
        const float x2 = (lane < 16) ? lbuf[row][lane + 64] : NEGF;
        float m = fmaxf(fmaxf(x0, x1), x2);
#pragma unroll
        for (int o = 16; o > 0; o >>= 1) m = fmaxf(m, __shfl_xor(m, o, 32));
        float s = __expf(x0 - m) + __expf(x1 - m) +
                  ((lane < 16) ? __expf(x2 - m) : 0.0f);
#pragma unroll
        for (int o = 16; o > 0; o >>= 1) s += __shfl_xor(s, o, 32);
        const float lse = m + __logf(s);
        float* orow = LP + (size_t)(tileM + row) * VOCAB;
        orow[lane] = x0 - lse;
        orow[lane + 32] = x1 - lse;
        if (lane < 16) orow[lane + 64] = x2 - lse;
    }
}

// ---------------------------------------------------------------- CTC DP
__device__ inline float lae(float a, float b) {
    const float m = fmaxf(a, b);
    const float n = fminf(a, b);
    return m + log1pf(__expf(n - m));
}

// One block per batch element; 401 extended states in LDS; sequential scan.
__global__ __launch_bounds__(512) void ctc_forward(
        const float* __restrict__ LP,       // [B][T][V] log-probs
        const int* __restrict__ targets,    // [B][LMAX]
        const int* __restrict__ dec_len,    // [B]
        const int* __restrict__ tgt_len,    // [B]
        float* __restrict__ out) {
    __shared__ int ext[2 * LMAX + 1];
    __shared__ unsigned char skip[2 * LMAX + 1];
    __shared__ float abuf[2][2 * LMAX + 1];
    const int S = 2 * LMAX + 1;             // 401
    const int b = blockIdx.x;
    const int tid = threadIdx.x;
    const int* tgt = targets + b * LMAX;
    const int Tb = dec_len[b];              // uniform within block
    const float* lp = LP + (size_t)b * TLEN * VOCAB;

    if (tid < S) {
        const int lab = (tid & 1) ? tgt[tid >> 1] : BLANK;
        ext[tid] = lab;
        unsigned char sk = 0;
        if ((tid & 1) && tid >= 3) sk = (lab != tgt[(tid >> 1) - 1]);
        skip[tid] = sk;
        float a = NEGF;
        if (tid == 0) a = lp[BLANK];
        else if (tid == 1) a = lp[tgt[0]];
        abuf[0][tid] = a;
    }
    __syncthreads();

    int cur = 0;
    for (int t = 1; t < TLEN; ++t) {
        if (t >= Tb) break;                 // uniform exit
        if (tid < S) {
            const float a0 = abuf[cur][tid];
            const float a1 = (tid >= 1) ? abuf[cur][tid - 1] : NEGF;
            const float a2 = skip[tid] ? abuf[cur][tid - 2] : NEGF;
            const float e = lp[t * VOCAB + ext[tid]];
            abuf[cur ^ 1][tid] = lae(lae(a0, a1), a2) + e;
        }
        __syncthreads();
        cur ^= 1;
    }

    if (tid == 0) {
        const int L = tgt_len[b];
        const float last = lae(abuf[cur][2 * L - 1], abuf[cur][2 * L]);
        const float loss = (last > NEGF * 0.5f) ? -last : 0.0f;
        atomicAdd(out, loss / (float)Tb * (1.0f / (float)BATCH));
    }
}

// ---------------------------------------------------------------- launcher
extern "C" void kernel_launch(void* const* d_in, const int* in_sizes, int n_in,
                              void* d_out, int out_size, void* d_ws, size_t ws_size,
                              hipStream_t stream) {
    const float* dec  = (const float*)d_in[0];
    const int*   tgt  = (const int*)d_in[1];
    const int*   dlen = (const int*)d_in[2];
    const int*   tlen = (const int*)d_in[3];
    const float* W1   = (const float*)d_in[4];
    const float* b1   = (const float*)d_in[5];
    const float* W2   = (const float*)d_in[6];
    const float* b2   = (const float*)d_in[7];
    float* out = (float*)d_out;

    char* ws = (char*)d_ws;
    size_t off = 0;
    auto carve = [&](size_t bytes) -> char* {
        char* p = ws + off;
        off += (bytes + 255) & ~(size_t)255;
        return p;
    };
    _Float16* Xh  = (_Float16*)carve((size_t)MROWS * DMODEL * 2);   // 65.5 MB
    _Float16* W1t = (_Float16*)carve((size_t)DMODEL * DMODEL * 2);  //  2.1 MB
    _Float16* W2t = (_Float16*)carve((size_t)VOCAB * DMODEL * 2);   //  0.2 MB
    _Float16* Hh  = (_Float16*)carve((size_t)MROWS * DMODEL * 2);   // 65.5 MB
    float*    LP  = (float*)   carve((size_t)MROWS * VOCAB * 4);    // 10.2 MB

    zero_out_kernel<<<1, 32, 0, stream>>>(out);

    transpose_convert_x<<<dim3((TLEN + 31) / 32, DMODEL / 32, BATCH),
                          256, 0, stream>>>(dec, Xh);
    convert_w1<<<(DMODEL * DMODEL) / 256, 256, 0, stream>>>(W1, W1t);
    convert_w2<<<(VOCAB * DMODEL + 255) / 256, 256, 0, stream>>>(W2, W2t);

    gemm1_wmma<<<dim3(MROWS / 16, 2), 256, 0, stream>>>(Xh, W1t, b1, Hh);
    gemm2_softmax_wmma<<<dim3(MROWS / 16), 160, 0, stream>>>(Hh, W2t, b2, LP);

    ctc_forward<<<BATCH, 512, 0, stream>>>(LP, tgt, dlen, tlen, out);
}
